// PixelLevelModeling_48868137894387
// MI455X (gfx1250) — matrix operand order repaired and checked by reading, not compiled, not run on hardware
//
#include <hip/hip_runtime.h>
#include <hip/hip_bf16.h>

// ---------------------------------------------------------------------------
// 3D Mamba (PixelLevelModeling) for gfx1250 / MI455X.
// Pipeline:
//   K0 prep_weights : fold dt_proj@x_proj[0:8] into 256x256; build padded
//                     [delta|B|C] weight (320x256); bf16-convert all GEMM wts
//   K1 ln_reshuffle : patch-shuffle + LayerNorm -> xn bf16 (128000 x 128)
//   K2 gemm (WMMA)  : xz = xn @ in_proj^T            (128000 x 512) bf16
//   K3 conv_silu    : causal depthwise conv(4) + silu -> u bf16
//   K4 gemm (WMMA)  : dbc = u @ Wd^T                 (128000 x 320) bf16
//   K5 scan         : selective scan, h[8] in regs, double-buffered LDS
//                     staging via async-to-LDS, fused +u*D and *silu(z)
//   K6 gemm (WMMA)  : o = yg @ out_proj^T            (128000 x 128) f32
//   K7 finalize     : un-shuffle + residual add -> d_out
// ---------------------------------------------------------------------------

#define NBATCH 2
#define CMODEL 128
#define ZDIM 40
#define DINNER 256
#define DSTATE 8
#define NSEQ 16           // NBATCH * 2*2*2
#define LSEQ 8000         // 20*20*20
#define MROWS 128000      // NSEQ * LSEQ
#define NPROJ 512         // 2*d_inner
#define NDBC 320          // 256 (delta) + 8 (B) + 8 (C) + 48 pad

typedef __attribute__((ext_vector_type(16))) __bf16 v16bf;
typedef __attribute__((ext_vector_type(8)))  __bf16 v8bf;
typedef __attribute__((ext_vector_type(4)))  __bf16 v4bf;
typedef __attribute__((ext_vector_type(8)))  float  v8f;
typedef __attribute__((ext_vector_type(2)))  int    v2i;

union BF16x16 { v16bf v; v8bf h[2]; };

// ---- gfx1250 async global->LDS copy (ASYNCcnt-tracked), feature-guarded ----
#if __has_builtin(__builtin_amdgcn_global_load_async_to_lds_b64)
#define HAVE_ASYNC_LDS 1
#endif

#if defined(HAVE_ASYNC_LDS)
// Probe-derived signature: (v2i addrspace(1)*, v2i addrspace(3)*, imm, imm).
// generic->AS casts go through integers (generic LDS addresses carry the LDS
// byte offset in the low 32 bits; global generic == addrspace(1) value).
#define CP8(gsrc, ldst)                                                        \
  __builtin_amdgcn_global_load_async_to_lds_b64(                               \
      (__attribute__((address_space(1))) v2i*)(unsigned long long)(gsrc),      \
      (__attribute__((address_space(3))) v2i*)(unsigned int)(unsigned long long)(ldst), \
      0, 0)
#if __has_builtin(__builtin_amdgcn_s_wait_asynccnt)
#define WAIT_ASYNC() __builtin_amdgcn_s_wait_asynccnt(0)
#else
#define WAIT_ASYNC() asm volatile("s_wait_asynccnt 0" ::: "memory")
#endif
#else
#define CP8(gsrc, ldst) (*(v4bf*)(ldst) = *(const v4bf*)(gsrc))
#define WAIT_ASYNC() ((void)0)
#endif

// A-fragment (16x32 bf16, MxK): lane = {M=lane&15, Kbase=(lane>>4)*16},
// holds K = base+[0..15]  -> one contiguous 32B run (two 16B chunks).
__device__ __forceinline__ v16bf load_a_frag(const __bf16* p) {
  BF16x16 r;
  r.h[0] = *(const v8bf*)(p);
  r.h[1] = *(const v8bf*)(p + 8);
  return r.v;
}
// B-fragment (32x16 bf16, KxN): lane = {N=lane&15, Kbase=(lane>>4)*16},
// holds K = base+[0..15]  -> one contiguous 32B run (two 16B chunks).
__device__ __forceinline__ v16bf load_b_frag(const __bf16* p) {
  BF16x16 r;
  r.h[0] = *(const v8bf*)(p);
  r.h[1] = *(const v8bf*)(p + 8);
  return r.v;
}

// ---------------------------------------------------------------------------
// Generic C[M,N] = A[M,K] @ W[N,K]^T, bf16 in, f32 accumulate via WMMA.
// Block = 256 threads = 8 waves. Wave -> 16(M) x NT*16(N) strip (NT wmma
// tiles, A-fragment reused NT times). grid = (M/128, N/(16*NT)). K mult of 32.
// ---------------------------------------------------------------------------
template <int NT, typename OutT>
__global__ void __launch_bounds__(256)
gemm_wmma_kernel(const __bf16* __restrict__ A, const __bf16* __restrict__ W,
                 OutT* __restrict__ Cout, int M, int N, int K) {
  const int lane = threadIdx.x & 31;
  const int wave = threadIdx.x >> 5;
  const int m0 = (blockIdx.x * 8 + wave) * 16;
  const int n_base = blockIdx.y * (NT * 16);
  const int rm = lane & 15;
  const int g = lane >> 4;

  v8f acc[NT] = {};
  const __bf16* arow = A + (size_t)(m0 + rm) * K + g * 16;
  const __bf16* wrow[NT];
#pragma unroll
  for (int t = 0; t < NT; ++t)
    wrow[t] = W + (size_t)(n_base + t * 16 + rm) * K + g * 16;

  for (int k0 = 0; k0 < K; k0 += 32) {
    __builtin_prefetch(arow + k0 + 128, 0, 1);   // global_prefetch_b8
    const v16bf a = load_a_frag(arow + k0);
#pragma unroll
    for (int t = 0; t < NT; ++t) {
      const v16bf b = load_b_frag(wrow[t] + k0);
      acc[t] = __builtin_amdgcn_wmma_f32_16x16x32_bf16(false, a, false, b,
                                                       (short)0, acc[t],
                                                       false, false);
    }
  }

  // D layout: VGPR r, lanes 0-15 -> M=r, lanes 16-31 -> M=8+r; N = lane&15.
#pragma unroll
  for (int t = 0; t < NT; ++t) {
#pragma unroll
    for (int r = 0; r < 8; ++r) {
      const int m = m0 + g * 8 + r;
      Cout[(size_t)m * N + n_base + t * 16 + rm] = (OutT)acc[t][r];
    }
  }
}

// ---------------------------------------------------------------------------
// K0: weight prep. Folds dt_proj (256x8) @ x_proj[0:8,:] (8x256) into a
// 256x256 combined delta weight; appends x_proj rows 8..23 (B,C proj),
// zero-pads to 320 rows. Also bf16-converts in/out projection weights.
// ---------------------------------------------------------------------------
__global__ void prep_weights(const float* __restrict__ in_proj_w,
                             const float* __restrict__ x_proj_w,
                             const float* __restrict__ dt_proj_w,
                             const float* __restrict__ out_proj_w,
                             __bf16* __restrict__ w_in, __bf16* __restrict__ wd,
                             __bf16* __restrict__ w_out) {
  const int tid = threadIdx.x;
  for (int i = tid; i < NPROJ * CMODEL; i += 256) w_in[i] = (__bf16)in_proj_w[i];
  for (int i = tid; i < CMODEL * DINNER; i += 256) w_out[i] = (__bf16)out_proj_w[i];
  for (int i = tid; i < DINNER * DINNER; i += 256) {
    const int e = i >> 8, d = i & 255;
    float s = 0.f;
#pragma unroll
    for (int r = 0; r < 8; ++r) s += dt_proj_w[e * 8 + r] * x_proj_w[r * DINNER + d];
    wd[(size_t)e * DINNER + d] = (__bf16)s;
  }
  for (int i = tid; i < 16 * DINNER; i += 256) {
    const int e = i >> 8, d = i & 255;
    wd[(size_t)(DINNER + e) * DINNER + d] = (__bf16)x_proj_w[(8 + e) * DINNER + d];
  }
  for (int i = tid; i < 48 * DINNER; i += 256)
    wd[(size_t)272 * DINNER + i] = (__bf16)0.f;
}

// ---------------------------------------------------------------------------
// K1: patch reshuffle + LayerNorm(C=128) -> xn bf16 [MROWS x 128].
// Block handles 64 (nb,l) positions staged in LDS.
// ---------------------------------------------------------------------------
__global__ void __launch_bounds__(256)
ln_reshuffle(const float* __restrict__ x, const float* __restrict__ gamma,
             const float* __restrict__ beta, __bf16* __restrict__ xn) {
  __shared__ float sx[64][130];
  __shared__ float smu[64], srs[64];
  const int tid = threadIdx.x;
  const int q0 = blockIdx.x * 64;

  for (int it = 0; it < 32; ++it) {
    const int idx = it * 256 + tid;
    const int pos = idx & 63;
    const int c = idx >> 6;
    const int q = q0 + pos;
    const int nb = q / LSEQ;
    const int l = q % LSEQ;
    const int b = nb >> 3, pz = (nb >> 2) & 1, ph = (nb >> 1) & 1, pw = nb & 1;
    const int nz = l / 400, r2 = l % 400, nh = r2 / 20, nw = r2 % 20;
    const int z = 2 * nz + pz, h = 2 * nh + ph, w = 2 * nw + pw;
    sx[pos][c] = x[((((size_t)b * CMODEL + c) * ZDIM + z) * ZDIM + h) * ZDIM + w];
  }
  __syncthreads();
  if (tid < 64) {
    float s = 0.f, s2 = 0.f;
    for (int c = 0; c < CMODEL; ++c) { float v = sx[tid][c]; s += v; s2 += v * v; }
    const float mu = s * (1.f / CMODEL);
    const float var = s2 * (1.f / CMODEL) - mu * mu;
    smu[tid] = mu;
    srs[tid] = rsqrtf(var + 1e-5f);
  }
  __syncthreads();
  for (int it = 0; it < 32; ++it) {
    const int idx = it * 256 + tid;
    const int c = idx & 127;
    const int pos = idx >> 7;
    const float v = (sx[pos][c] - smu[pos]) * srs[pos] * gamma[c] + beta[c];
    xn[(size_t)(q0 + pos) * CMODEL + c] = (__bf16)v;
  }
}

// ---------------------------------------------------------------------------
// K3: causal depthwise conv (k=4) over xp = xz[:, :256], then silu -> u.
// ---------------------------------------------------------------------------
__global__ void __launch_bounds__(256)
conv_silu(const __bf16* __restrict__ xz, const float* __restrict__ conv_w,
          const float* __restrict__ conv_b, __bf16* __restrict__ u) {
  const long long gid = (long long)blockIdx.x * 256 + threadIdx.x;
  const int d = (int)(gid & 255);
  const int row = (int)(gid >> 8);
  const int nb = row / LSEQ;
  const int l = row % LSEQ;
  float acc = conv_b[d];
#pragma unroll
  for (int k = 0; k < 4; ++k) {
    const int ls = l - 3 + k;
    if (ls >= 0)
      acc += (float)xz[((size_t)nb * LSEQ + ls) * NPROJ + d] * conv_w[d * 4 + k];
  }
  const float s = acc / (1.f + __expf(-acc));
  u[(size_t)row * DINNER + d] = (__bf16)s;
}

// ---------------------------------------------------------------------------
// K5: selective scan. 64 blocks x 64 threads: block = (nb, 64-channel slice),
// thread = one channel, h[8] in registers. Chunks of 32 steps of delta/u/z/B/C
// double-buffered through LDS with async-to-LDS prefetch of the next chunk
// overlapping the sequential recurrence. Fused softplus, +u*D, *silu(z).
// ---------------------------------------------------------------------------
__global__ void __launch_bounds__(64)
scan_kernel(const __bf16* __restrict__ dbc, const __bf16* __restrict__ u,
            const __bf16* __restrict__ xz, const float* __restrict__ dt_bias,
            const float* __restrict__ A_log, const float* __restrict__ D_vec,
            __bf16* __restrict__ yg) {
  const int tid = threadIdx.x;
  const int nb = blockIdx.x >> 2;
  const int dbase = (blockIdx.x & 3) * 64;
  const int d = dbase + tid;

  __shared__ __bf16 sdel[2][32][64];
  __shared__ __bf16 su[2][32][64];
  __shared__ __bf16 sz[2][32][64];
  __shared__ __bf16 sbc[2][32][16];   // [0..7]=B_t, [8..15]=C_t

  float Acoef[DSTATE], h[DSTATE];
#pragma unroll
  for (int s = 0; s < DSTATE; ++s) {
    Acoef[s] = -__expf(A_log[d * DSTATE + s]);
    h[s] = 0.f;
  }
  const float bias = dt_bias[d];
  const float Dv = D_vec[d];

  // Stage one 32-step chunk into LDS buffer `buf` (all 64 threads, 8B units).
  auto stage = [&](int buf, int c0) {
    const size_t rowbase = (size_t)nb * LSEQ + c0;
#pragma unroll
    for (int it = 0; it < 8; ++it) {
      const int idx = it * 64 + tid;
      const int i = idx >> 4;
      const int cg = (idx & 15) << 2;
      CP8(&dbc[(rowbase + i) * NDBC + dbase + cg], &sdel[buf][i][cg]);
      CP8(&u[(rowbase + i) * DINNER + dbase + cg], &su[buf][i][cg]);
      CP8(&xz[(rowbase + i) * NPROJ + DINNER + dbase + cg], &sz[buf][i][cg]);
    }
#pragma unroll
    for (int it = 0; it < 2; ++it) {
      const int idx = it * 64 + tid;
      const int i = idx >> 2;
      const int cg = (idx & 3) << 2;
      CP8(&dbc[(rowbase + i) * NDBC + DINNER + cg], &sbc[buf][i][cg]);
    }
  };

  stage(0, 0);
  int buf = 0;
  for (int c0 = 0; c0 < LSEQ; c0 += 32) {
    WAIT_ASYNC();          // my wave's staged loads landed in LDS
    __syncthreads();       // everyone's landed; prev chunk fully consumed
    if (c0 + 32 < LSEQ) stage(buf ^ 1, c0 + 32);   // overlap next chunk

    const size_t rowbase = (size_t)nb * LSEQ + c0;
    for (int i = 0; i < 32; ++i) {
      const float dp = (float)sdel[buf][i][tid] + bias;
      const float delta = dp > 20.f ? dp : log1pf(__expf(dp));
      const float ut = (float)su[buf][i][tid];
      const float du = delta * ut;
      float y = 0.f;
#pragma unroll
      for (int s = 0; s < DSTATE; ++s) {
        h[s] = __expf(delta * Acoef[s]) * h[s] + du * (float)sbc[buf][i][s];
        y += h[s] * (float)sbc[buf][i][8 + s];
      }
      y += ut * Dv;
      const float zv = (float)sz[buf][i][tid];
      const float gate = zv / (1.f + __expf(-zv));
      yg[(rowbase + i) * DINNER + d] = (__bf16)(y * gate);
    }
    buf ^= 1;
  }
}

// ---------------------------------------------------------------------------
// K7: un-shuffle (NB,L,C) -> (B,C,Z,H,W) + residual add.
// ---------------------------------------------------------------------------
__global__ void __launch_bounds__(256)
finalize_kernel(const float* __restrict__ x, const float* __restrict__ o,
                float* __restrict__ out) {
  const size_t idx = (size_t)blockIdx.x * 256 + threadIdx.x;
  const int w = (int)(idx % 40);
  size_t t = idx / 40;
  const int h = (int)(t % 40); t /= 40;
  const int z = (int)(t % 40); t /= 40;
  const int c = (int)(t % CMODEL);
  const int b = (int)(t / CMODEL);
  const int nb = ((b * 2 + (z & 1)) * 2 + (h & 1)) * 2 + (w & 1);
  const int l = ((z >> 1) * 20 + (h >> 1)) * 20 + (w >> 1);
  out[idx] = x[idx] + o[((size_t)nb * LSEQ + l) * CMODEL + c];
}

// ---------------------------------------------------------------------------
extern "C" void kernel_launch(void* const* d_in, const int* in_sizes, int n_in,
                              void* d_out, int out_size, void* d_ws, size_t ws_size,
                              hipStream_t stream) {
  (void)in_sizes; (void)n_in; (void)out_size; (void)ws_size;
  const float* x = (const float*)d_in[0];
  const float* norm_g = (const float*)d_in[1];
  const float* norm_b = (const float*)d_in[2];
  const float* in_proj_w = (const float*)d_in[3];
  const float* conv_w = (const float*)d_in[4];
  const float* conv_b = (const float*)d_in[5];
  const float* x_proj_w = (const float*)d_in[6];
  const float* dt_proj_w = (const float*)d_in[7];
  const float* dt_bias = (const float*)d_in[8];
  const float* A_log = (const float*)d_in[9];
  const float* D_vec = (const float*)d_in[10];
  const float* out_proj_w = (const float*)d_in[11];
  float* out = (float*)d_out;

  char* p = (char*)d_ws;
  auto alloc = [&](size_t bytes) -> char* {
    char* r = p;
    p += (bytes + 255) & ~(size_t)255;
    return r;
  };
  __bf16* w_in = (__bf16*)alloc((size_t)NPROJ * CMODEL * 2);
  __bf16* wd = (__bf16*)alloc((size_t)NDBC * DINNER * 2);
  __bf16* w_out = (__bf16*)alloc((size_t)CMODEL * DINNER * 2);
  __bf16* xn = (__bf16*)alloc((size_t)MROWS * CMODEL * 2);
  __bf16* xz = (__bf16*)alloc((size_t)MROWS * NPROJ * 2);
  __bf16* ubuf = (__bf16*)alloc((size_t)MROWS * DINNER * 2);
  __bf16* dbc = (__bf16*)alloc((size_t)MROWS * NDBC * 2);
  __bf16* yg = (__bf16*)alloc((size_t)MROWS * DINNER * 2);
  float* o_tmp = (float*)alloc((size_t)MROWS * CMODEL * 4);

  prep_weights<<<1, 256, 0, stream>>>(in_proj_w, x_proj_w, dt_proj_w, out_proj_w,
                                      w_in, wd, w_out);
  ln_reshuffle<<<MROWS / 64, 256, 0, stream>>>(x, norm_g, norm_b, xn);
  // N=512: NT=8 (wave strip 16x128) halves A-stream re-reads vs NT=4.
  gemm_wmma_kernel<8, __bf16><<<dim3(MROWS / 128, NPROJ / 128), 256, 0, stream>>>(
      xn, w_in, xz, MROWS, NPROJ, CMODEL);
  conv_silu<<<MROWS * DINNER / 256, 256, 0, stream>>>(xz, conv_w, conv_b, ubuf);
  gemm_wmma_kernel<4, __bf16><<<dim3(MROWS / 128, NDBC / 64), 256, 0, stream>>>(
      ubuf, wd, dbc, MROWS, NDBC, DINNER);
  scan_kernel<<<NSEQ * 4, 64, 0, stream>>>(dbc, ubuf, xz, dt_bias, A_log, D_vec, yg);
  gemm_wmma_kernel<4, float><<<dim3(MROWS / 128, CMODEL / 64), 256, 0, stream>>>(
      yg, w_out, o_tmp, MROWS, CMODEL, DINNER);
  finalize_kernel<<<(NBATCH * CMODEL * ZDIM * ZDIM * ZDIM) / 256, 256, 0, stream>>>(
      x, o_tmp, out);
}
